// Model_54606214202234
// MI455X (gfx1250) — compile-verified, matrix operand order checked by reading, not run on hardware
//
#include <hip/hip_runtime.h>
#include <hip/hip_bf16.h>
#include <math.h>

// ---------------------------------------------------------------------------
// Informer encoder forward for MI455X (gfx1250, wave32).
// GEMMs: v_wmma_f32_16x16x32_bf16, LDS-staged 128x64 block tiles,
// double-buffered with ASYNCcnt-tracked global->LDS async copies when the
// toolchain exposes them (fallback: regular loads + ds_store staging).
// Activations are converted to bf16 once per GEMM input (not per re-read).
// ---------------------------------------------------------------------------

typedef __attribute__((ext_vector_type(16))) __bf16 v16bf;
typedef __attribute__((ext_vector_type(8)))  __bf16 v8bf;
typedef __attribute__((ext_vector_type(8)))  float  v8f;

#define DM   512
#define DFF  2048
#define BATCH 4
#define SEQL 2048
#define HEADS 8
#define DK   64
#define NL   2
#define UTOP 40
#define NGRP 100

// ----- async global->LDS copy (gfx1250 ASYNCcnt path), guarded -------------
#if defined(__has_builtin)
#if __has_builtin(__builtin_amdgcn_global_load_async_to_lds_b128) && \
    __has_builtin(__builtin_amdgcn_s_wait_asynccnt)
#define USE_ASYNC_COPY 1
#endif
#endif
#ifndef USE_ASYNC_COPY
#define USE_ASYNC_COPY 0
#endif

#if USE_ASYNC_COPY
typedef int v4i_vs __attribute__((vector_size(16)));
typedef __attribute__((address_space(1))) v4i_vs GlobalV4;
typedef __attribute__((address_space(3))) v4i_vs LocalV4;
#endif

__device__ inline void copy16_g2l(const __bf16* g, __bf16* l) {
#if USE_ASYNC_COPY
  GlobalV4* gp = (GlobalV4*)(unsigned long long)(size_t)g;
  LocalV4*  lp = (LocalV4*)(unsigned int)(size_t)l;
  __builtin_amdgcn_global_load_async_to_lds_b128(gp, lp, 0, 0);
#else
  *(uint4*)l = *(const uint4*)g;
#endif
}

__device__ inline void wait_async_copies() {
#if USE_ASYNC_COPY
  __builtin_amdgcn_s_wait_asynccnt(0);
#endif
}

// -------------------- fp32 -> bf16 conversion pass -------------------------
__global__ void __launch_bounds__(256)
cvt_bf16_kernel(const float* __restrict__ in, __bf16* __restrict__ out, size_t n) {
  size_t gid = (size_t)blockIdx.x * 256 + threadIdx.x;
  if (gid < n) out[gid] = (__bf16)in[gid];
}

// -------------------- weight pack: fp32 (K x N) -> bf16 (N x K) ------------
__global__ void __launch_bounds__(256)
pack_w_kernel(const float* __restrict__ W, __bf16* __restrict__ Wt, int K, int N) {
  size_t gid = (size_t)blockIdx.x * 256 + threadIdx.x;
  if (gid >= (size_t)K * N) return;
  int n = (int)(gid % N);
  int k = (int)(gid / N);
  Wt[(size_t)n * K + k] = (__bf16)W[(size_t)k * N + n];
}

// -------------------- WMMA GEMM with LDS staging ---------------------------
// Block = 256 threads (8 waves). Block tile: 128 (M) x 64 (N).
// Wave w owns rows [w*16, w*16+16); computes 4 16x16 subtiles along N.
// K processed in 32-wide tiles, double-buffered in LDS.

#define MBLK 128
#define NBLK 64
#define KTILE 32

__device__ inline void load_tile(const __bf16* __restrict__ Abase,
                                 const __bf16* __restrict__ Bbase, int K, int kt,
                                 __bf16* __restrict__ Alds,
                                 __bf16* __restrict__ Blds, int tid) {
  // A tile: 128 rows x 32 K = 512 x 16B chunks
#pragma unroll
  for (int c = tid; c < (MBLK * KTILE) / 8; c += 256) {
    int row = c >> 2, ko = (c & 3) * 8;
    copy16_g2l(Abase + (size_t)row * K + kt * KTILE + ko, Alds + row * KTILE + ko);
  }
  // B tile: 64 rows x 32 K = 256 x 16B chunks
  {
    int row = tid >> 2, ko = (tid & 3) * 8;
    copy16_g2l(Bbase + (size_t)row * K + kt * KTILE + ko, Blds + row * KTILE + ko);
  }
}

__device__ inline v16bf frag_from_lds(const __bf16* __restrict__ row, int lh) {
  v8bf a0 = *(const v8bf*)(row + lh * 8);
  v8bf a1 = *(const v8bf*)(row + 16 + lh * 8);
  return __builtin_shufflevector(a0, a1, 0,1,2,3,4,5,6,7,8,9,10,11,12,13,14,15);
}

__global__ void __launch_bounds__(256)
gemm_wmma_bf16(const __bf16* __restrict__ A, const __bf16* __restrict__ Wt,
               const float* __restrict__ bias, const float* __restrict__ resid,
               float* __restrict__ C, __bf16* __restrict__ Cbf,
               int M, int N, int K, int fuse_gelu) {
  __shared__ __bf16 Alds[2][MBLK * KTILE];
  __shared__ __bf16 Blds[2][NBLK * KTILE];

  int tid = threadIdx.x;
  int w = tid >> 5, lane = tid & 31, lm = lane & 15, lh = lane >> 4;
  int nblkN = N / NBLK;
  int bm = blockIdx.x / nblkN, bn = blockIdx.x % nblkN;
  const __bf16* Abase = A  + (size_t)bm * MBLK * K;
  const __bf16* Bbase = Wt + (size_t)bn * NBLK * K;
  const int NT = K / KTILE;

  v8f acc[4] = {v8f{}, v8f{}, v8f{}, v8f{}};

  load_tile(Abase, Bbase, K, 0, Alds[0], Blds[0], tid);
  wait_async_copies();
  __syncthreads();

  for (int kt = 0; kt < NT; ++kt) {
    int cur = kt & 1;
    if (kt + 1 < NT)
      load_tile(Abase, Bbase, K, kt + 1, Alds[cur ^ 1], Blds[cur ^ 1], tid);

    const __bf16* Ar = Alds[cur] + (w * 16 + lm) * KTILE;
    v16bf a = frag_from_lds(Ar, lh);
#pragma unroll
    for (int s = 0; s < 4; ++s) {
      const __bf16* Br = Blds[cur] + (s * 16 + lm) * KTILE;
      v16bf b = frag_from_lds(Br, lh);
      acc[s] = __builtin_amdgcn_wmma_f32_16x16x32_bf16(
          /*neg_a=*/false, a, /*neg_b=*/false, b,
          /*c_mod=*/(short)0, acc[s], /*reuse_a=*/false, /*reuse_b=*/false);
    }

    wait_async_copies();
    __syncthreads();
  }

  // epilogue: bias (+gelu) (+residual); fp32 or bf16 output
  int nbase = bn * NBLK + lm;
  int mbase = bm * MBLK + w * 16 + lh * 8;
#pragma unroll
  for (int s = 0; s < 4; ++s) {
    int n = nbase + s * 16;
    float bv = bias ? bias[n] : 0.0f;
#pragma unroll
    for (int i = 0; i < 8; ++i) {
      int m = mbase + i;
      float v = acc[s][i] + bv;
      if (fuse_gelu) v = 0.5f * v * (1.0f + erff(v * 0.70710678118654752f));
      if (resid) v += resid[(size_t)m * N + n];
      if (Cbf) Cbf[(size_t)m * N + n] = (__bf16)v;
      else     C[(size_t)m * N + n] = v;
    }
  }
}

// -------------------- embedding: emb + circular conv + time + PE -----------
__global__ void __launch_bounds__(256)
embed_kernel(const int* __restrict__ st, const float* __restrict__ xenc,
             const float* __restrict__ xmark, const float* __restrict__ embW,
             const float* __restrict__ convW, const float* __restrict__ timeW,
             const float* __restrict__ timeb, float* __restrict__ x) {
  size_t gid = (size_t)blockIdx.x * 256 + threadIdx.x;
  size_t total = (size_t)BATCH * SEQL * DM;
  if (gid >= total) return;
  int o = (int)(gid % DM);
  size_t bl = gid / DM;
  int l = (int)(bl % SEQL);
  int b = (int)(bl / SEQL);

  float v = embW[(size_t)st[bl] * DM + o];

  const float* wrow = convW + (size_t)o * 21;   // (C_IN=7, T=3)
#pragma unroll
  for (int t = 0; t < 3; ++t) {
    int ls = l + t - 1;
    ls = (ls < 0) ? ls + SEQL : (ls >= SEQL ? ls - SEQL : ls);
    const float* xe = xenc + ((size_t)b * SEQL + ls) * 7;
#pragma unroll
    for (int c = 0; c < 7; ++c) v += wrow[c * 3 + t] * xe[c];
  }

  const float* xm = xmark + bl * 4;
#pragma unroll
  for (int mm = 0; mm < 4; ++mm) v += xm[mm] * timeW[mm * DM + o];
  v += timeb[o];

  float div = __expf(-(float)(o & ~1) * (9.210340371976184f / (float)DM));
  float ang = (float)l * div;
  v += (o & 1) ? __cosf(ang) : __sinf(ang);
  x[gid] = v;
}

// -------------------- sparse-score statistic M = max - mean ----------------
__device__ inline int sample_idx(int l, int j) {
  unsigned s = (unsigned)(l * UTOP + j) * 0x9E3779B1u;
  s ^= s >> 16; s *= 0x7feb352du; s ^= s >> 15; s *= 0x846ca68bu; s ^= s >> 16;
  return (int)(s & (SEQL - 1));
}

__global__ void __launch_bounds__(256)
attn_m_kernel(const float* __restrict__ q, const float* __restrict__ k,
              float* __restrict__ Mbuf) {
  int wid  = blockIdx.x * 8 + (threadIdx.x >> 5);
  int lane = threadIdx.x & 31;
  int total = BATCH * HEADS * SEQL;
  if (wid >= total) return;
  int l  = wid % SEQL;
  int bh = wid / SEQL;
  int h = bh % HEADS, b = bh / HEADS;
  const float* qrow = q + ((size_t)b * SEQL + l) * DM + h * DK;
  const float* kb   = k + ((size_t)b * SEQL) * DM + h * DK;
  float mx = -3.4e38f, sm = 0.0f;
  for (int j = lane; j < UTOP; j += 32) {
    const float* krow = kb + (size_t)sample_idx(l, j) * DM;
    float d = 0.0f;
#pragma unroll
    for (int t = 0; t < DK; ++t) d += qrow[t] * krow[t];
    mx = fmaxf(mx, d);
    sm += d;
  }
#pragma unroll
  for (int o = 16; o > 0; o >>= 1) {
    mx = fmaxf(mx, __shfl_xor(mx, o, 32));
    sm += __shfl_xor(sm, o, 32);
  }
  if (lane == 0) Mbuf[wid] = mx - sm / (float)UTOP;
}

// -------------------- iterative top-k (u = 40) per (b,h) -------------------
__global__ void __launch_bounds__(256)
topk_kernel(float* __restrict__ Mbuf, int* __restrict__ Mtop) {
  int bh = blockIdx.x;
  float* Mrow = Mbuf + (size_t)bh * SEQL;
  __shared__ float sv[256];
  __shared__ int   si[256];
  for (int t = 0; t < UTOP; ++t) {
    float bv = -3.4e38f; int bi = 0;
    for (int i = threadIdx.x; i < SEQL; i += 256) {
      float v = Mrow[i];
      if (v > bv) { bv = v; bi = i; }
    }
    sv[threadIdx.x] = bv; si[threadIdx.x] = bi;
    __syncthreads();
    for (int o = 128; o > 0; o >>= 1) {
      if (threadIdx.x < o && sv[threadIdx.x + o] > sv[threadIdx.x]) {
        sv[threadIdx.x] = sv[threadIdx.x + o];
        si[threadIdx.x] = si[threadIdx.x + o];
      }
      __syncthreads();
    }
    if (threadIdx.x == 0) {
      Mtop[bh * UTOP + t] = si[0];
      Mrow[si[0]] = -3.4e38f;
    }
    __syncthreads();
  }
}

// -------------------- scores + softmax + update + V-mean -------------------
__global__ void __launch_bounds__(256)
attn_ctx_kernel(const float* __restrict__ q, const float* __restrict__ k,
                const float* __restrict__ v, const int* __restrict__ Mtop,
                float* __restrict__ scores, float* __restrict__ vmean,
                float* __restrict__ upd) {
  int bh = blockIdx.x;
  int h = bh % HEADS, b = bh / HEADS;
  int tid = threadIdx.x;
  __shared__ float sm[256];
  const float* vb = v + ((size_t)b * SEQL) * DM + h * DK;
  const float* kb = k + ((size_t)b * SEQL) * DM + h * DK;
  const float* qb = q + ((size_t)b * SEQL) * DM + h * DK;

  // V mean over L
  {
    int d = tid & 63, part = tid >> 6;
    float s = 0.0f;
    for (int l = part * (SEQL / 4); l < (part + 1) * (SEQL / 4); ++l)
      s += vb[(size_t)l * DM + d];
    sm[tid] = s;
    __syncthreads();
    if (tid < 64)
      vmean[(size_t)bh * DK + tid] =
          (sm[tid] + sm[tid + 64] + sm[tid + 128] + sm[tid + 192]) / (float)SEQL;
    __syncthreads();
  }

  // scores (u x L), scaled
  float* sc = scores + (size_t)bh * UTOP * SEQL;
  const float scale = 0.125f;  // 1/sqrt(64)
  for (int idx = tid; idx < UTOP * SEQL; idx += 256) {
    int u = idx / SEQL, l = idx % SEQL;
    const float* qrow = qb + (size_t)Mtop[bh * UTOP + u] * DM;
    const float* krow = kb + (size_t)l * DM;
    float d = 0.0f;
#pragma unroll
    for (int t = 0; t < DK; ++t) d += qrow[t] * krow[t];
    sc[idx] = d * scale;
  }
  __syncthreads();

  // softmax per row (one warp per row)
  int w = tid >> 5, lane = tid & 31;
  for (int u = w; u < UTOP; u += 8) {
    float* row = sc + (size_t)u * SEQL;
    float mx = -3.4e38f;
    for (int l = lane; l < SEQL; l += 32) mx = fmaxf(mx, row[l]);
#pragma unroll
    for (int o = 16; o > 0; o >>= 1) mx = fmaxf(mx, __shfl_xor(mx, o, 32));
    float s = 0.0f;
    for (int l = lane; l < SEQL; l += 32) {
      float e = __expf(row[l] - mx);
      row[l] = e;
      s += e;
    }
#pragma unroll
    for (int o = 16; o > 0; o >>= 1) s += __shfl_xor(s, o, 32);
    float inv = 1.0f / s;
    for (int l = lane; l < SEQL; l += 32) row[l] *= inv;
  }
  __syncthreads();

  // upd = attn @ V
  for (int idx = tid; idx < UTOP * DK; idx += 256) {
    int u = idx / DK, d = idx % DK;
    const float* row = sc + (size_t)u * SEQL;
    float s = 0.0f;
    for (int l = 0; l < SEQL; ++l) s += row[l] * vb[(size_t)l * DM + d];
    upd[((size_t)bh * UTOP + u) * DK + d] = s;
  }
}

// -------------------- ctx = broadcast(Vmean); ctx[M_top] = upd -------------
__global__ void __launch_bounds__(256)
bcast_ctx_kernel(const float* __restrict__ vmean, float* __restrict__ ctx) {
  size_t gid = (size_t)blockIdx.x * 256 + threadIdx.x;
  size_t total = (size_t)BATCH * SEQL * DM;
  if (gid >= total) return;
  int o = (int)(gid % DM);
  size_t bl = gid / DM;
  int b = (int)(bl / SEQL);
  int h = o / DK, d = o % DK;
  ctx[gid] = vmean[((size_t)(b * HEADS + h)) * DK + d];
}

__global__ void __launch_bounds__(256)
scatter_ctx_kernel(const float* __restrict__ upd, const int* __restrict__ Mtop,
                   float* __restrict__ ctx) {
  size_t gid = (size_t)blockIdx.x * 256 + threadIdx.x;
  size_t total = (size_t)BATCH * HEADS * UTOP * DK;
  if (gid >= total) return;
  int d = (int)(gid % DK);
  int u = (int)((gid / DK) % UTOP);
  int bh = (int)(gid / ((size_t)DK * UTOP));
  int h = bh % HEADS, b = bh / HEADS;
  int l = Mtop[bh * UTOP + u];
  ctx[((size_t)b * SEQL + l) * DM + h * DK + d] = upd[gid];
}

// -------------------- LayerNorm (row of 512) -------------------------------
__global__ void __launch_bounds__(256)
ln_kernel(const float* __restrict__ in, const float* __restrict__ g,
          const float* __restrict__ b, float* __restrict__ out) {
  int row = blockIdx.x, tid = threadIdx.x;
  const float* x = in + (size_t)row * DM;
  __shared__ float red[256];
  float s = 0.0f;
  for (int i = tid; i < DM; i += 256) s += x[i];
  red[tid] = s; __syncthreads();
  for (int o = 128; o > 0; o >>= 1) { if (tid < o) red[tid] += red[tid + o]; __syncthreads(); }
  float mu = red[0] / (float)DM; __syncthreads();
  float vv = 0.0f;
  for (int i = tid; i < DM; i += 256) { float d = x[i] - mu; vv += d * d; }
  red[tid] = vv; __syncthreads();
  for (int o = 128; o > 0; o >>= 1) { if (tid < o) red[tid] += red[tid + o]; __syncthreads(); }
  float rstd = rsqrtf(red[0] / (float)DM + 1e-5f); __syncthreads();
  for (int i = tid; i < DM; i += 256)
    out[(size_t)row * DM + i] = (x[i] - mu) * rstd * g[i] + b[i];
}

// -------------------- final LN (last row) + logits -------------------------
__global__ void __launch_bounds__(256)
logits_kernel(const float* __restrict__ x, const float* __restrict__ g,
              const float* __restrict__ bb, const float* __restrict__ embW,
              float* __restrict__ out) {
  int b = blockIdx.x, tid = threadIdx.x;
  const float* row = x + ((size_t)b * SEQL + (SEQL - 1)) * DM;
  __shared__ float red[256];
  __shared__ float nx[DM];
  float s = 0.0f;
  for (int i = tid; i < DM; i += 256) s += row[i];
  red[tid] = s; __syncthreads();
  for (int o = 128; o > 0; o >>= 1) { if (tid < o) red[tid] += red[tid + o]; __syncthreads(); }
  float mu = red[0] / (float)DM; __syncthreads();
  float vv = 0.0f;
  for (int i = tid; i < DM; i += 256) { float d = row[i] - mu; vv += d * d; }
  red[tid] = vv; __syncthreads();
  for (int o = 128; o > 0; o >>= 1) { if (tid < o) red[tid] += red[tid + o]; __syncthreads(); }
  float rstd = rsqrtf(red[0] / (float)DM + 1e-5f); __syncthreads();
  for (int i = tid; i < DM; i += 256) nx[i] = (row[i] - mu) * rstd * g[i] + bb[i];
  __syncthreads();
  for (int gg = tid; gg < NGRP; gg += 256) {
    const float* w = embW + (size_t)gg * DM;
    float d = 0.0f;
#pragma unroll 8
    for (int t = 0; t < DM; ++t) d += nx[t] * w[t];
    out[b * NGRP + gg] = d;
  }
}

// ---------------------------------------------------------------------------
static inline size_t alignup(size_t v) { return (v + 255) & ~(size_t)255; }

extern "C" void kernel_launch(void* const* d_in, const int* in_sizes, int n_in,
                              void* d_out, int out_size, void* d_ws, size_t ws_size,
                              hipStream_t stream) {
  (void)in_sizes; (void)n_in; (void)out_size; (void)ws_size;

  const int*   x_state   = (const int*)d_in[0];
  const float* x_enc     = (const float*)d_in[1];
  const float* x_mark    = (const float*)d_in[2];
  const float* emb_W     = (const float*)d_in[3];
  const float* conv_W    = (const float*)d_in[4];
  const float* time_W    = (const float*)d_in[5];
  const float* time_b    = (const float*)d_in[6];
  const float* Wq        = (const float*)d_in[7];
  const float* bq        = (const float*)d_in[8];
  const float* Wk        = (const float*)d_in[9];
  const float* bk        = (const float*)d_in[10];
  const float* Wv        = (const float*)d_in[11];
  const float* bv        = (const float*)d_in[12];
  const float* Wo        = (const float*)d_in[13];
  const float* bo        = (const float*)d_in[14];
  const float* ln1_g     = (const float*)d_in[15];
  const float* ln1_b     = (const float*)d_in[16];
  const float* ln2_g     = (const float*)d_in[17];
  const float* ln2_b     = (const float*)d_in[18];
  const float* ffn1_W    = (const float*)d_in[19];
  const float* ffn1_b    = (const float*)d_in[20];
  const float* ffn2_W    = (const float*)d_in[21];
  const float* ffn2_b    = (const float*)d_in[22];
  const float* norm_g    = (const float*)d_in[23];
  const float* norm_b    = (const float*)d_in[24];
  float* out = (float*)d_out;

  const int M = BATCH * SEQL;         // 8192 rows
  char* ws = (char*)d_ws;
  size_t off = 0;
  float*  x    = (float*) (ws + off); off += alignup((size_t)M * DM * 4);
  float*  xt   = (float*) (ws + off); off += alignup((size_t)M * DM * 4);
  float*  q    = (float*) (ws + off); off += alignup((size_t)M * DM * 4);
  float*  k    = (float*) (ws + off); off += alignup((size_t)M * DM * 4);
  float*  v    = (float*) (ws + off); off += alignup((size_t)M * DM * 4);
  float*  ctx  = (float*) (ws + off); off += alignup((size_t)M * DM * 4);
  __bf16* xbf  = (__bf16*)(ws + off); off += alignup((size_t)M * DM * 2);
  __bf16* cbf  = (__bf16*)(ws + off); off += alignup((size_t)M * DM * 2);
  __bf16* y1bf = (__bf16*)(ws + off); off += alignup((size_t)M * DFF * 2);
  float*  Mbuf = (float*) (ws + off); off += alignup((size_t)BATCH * HEADS * SEQL * 4);
  int*    Mtop = (int*)   (ws + off); off += alignup((size_t)BATCH * HEADS * UTOP * 4);
  float*  sc   = (float*) (ws + off); off += alignup((size_t)BATCH * HEADS * UTOP * SEQL * 4);
  float*  vme  = (float*) (ws + off); off += alignup((size_t)BATCH * HEADS * DK * 4);
  float*  upd  = (float*) (ws + off); off += alignup((size_t)BATCH * HEADS * UTOP * DK * 4);
  __bf16* Wqt[NL], *Wkt[NL], *Wvt[NL], *Wot[NL], *F1t[NL], *F2t[NL];
  for (int l = 0; l < NL; ++l) {
    Wqt[l] = (__bf16*)(ws + off); off += alignup((size_t)DM * DM * 2);
    Wkt[l] = (__bf16*)(ws + off); off += alignup((size_t)DM * DM * 2);
    Wvt[l] = (__bf16*)(ws + off); off += alignup((size_t)DM * DM * 2);
    Wot[l] = (__bf16*)(ws + off); off += alignup((size_t)DM * DM * 2);
    F1t[l] = (__bf16*)(ws + off); off += alignup((size_t)DM * DFF * 2);
    F2t[l] = (__bf16*)(ws + off); off += alignup((size_t)DFF * DM * 2);
  }

  // ---- pack weights into bf16 transposed layouts ----
  const int pk_sq = (DM * DM) / 256, pk_ff = (DM * DFF) / 256;
  for (int l = 0; l < NL; ++l) {
    pack_w_kernel<<<pk_sq, 256, 0, stream>>>(Wq + (size_t)l * DM * DM, Wqt[l], DM, DM);
    pack_w_kernel<<<pk_sq, 256, 0, stream>>>(Wk + (size_t)l * DM * DM, Wkt[l], DM, DM);
    pack_w_kernel<<<pk_sq, 256, 0, stream>>>(Wv + (size_t)l * DM * DM, Wvt[l], DM, DM);
    pack_w_kernel<<<pk_sq, 256, 0, stream>>>(Wo + (size_t)l * DM * DM, Wot[l], DM, DM);
    pack_w_kernel<<<pk_ff, 256, 0, stream>>>(ffn1_W + (size_t)l * DM * DFF, F1t[l], DM, DFF);
    pack_w_kernel<<<pk_ff, 256, 0, stream>>>(ffn2_W + (size_t)l * DFF * DM, F2t[l], DFF, DM);
  }

  // ---- embedding ----
  embed_kernel<<<(M * DM) / 256, 256, 0, stream>>>(x_state, x_enc, x_mark, emb_W,
                                                   conv_W, time_W, time_b, x);

  const int cvt_dm_blocks  = (M * DM) / 256;
  const int gemm_dm_blocks  = (M / MBLK) * (DM / NBLK);    // 64*8  = 512
  const int gemm_dff_blocks = (M / MBLK) * (DFF / NBLK);   // 64*32 = 2048

  for (int l = 0; l < NL; ++l) {
    // QKV projections (WMMA, bf16 A staged through LDS)
    cvt_bf16_kernel<<<cvt_dm_blocks, 256, 0, stream>>>(x, xbf, (size_t)M * DM);
    gemm_wmma_bf16<<<gemm_dm_blocks, 256, 0, stream>>>(xbf, Wqt[l], bq + l * DM, nullptr, q, nullptr, M, DM, DM, 0);
    gemm_wmma_bf16<<<gemm_dm_blocks, 256, 0, stream>>>(xbf, Wkt[l], bk + l * DM, nullptr, k, nullptr, M, DM, DM, 0);
    gemm_wmma_bf16<<<gemm_dm_blocks, 256, 0, stream>>>(xbf, Wvt[l], bv + l * DM, nullptr, v, nullptr, M, DM, DM, 0);

    // ProbSparse attention
    attn_m_kernel<<<(BATCH * HEADS * SEQL) / 8, 256, 0, stream>>>(q, k, Mbuf);
    topk_kernel<<<BATCH * HEADS, 256, 0, stream>>>(Mbuf, Mtop);
    attn_ctx_kernel<<<BATCH * HEADS, 256, 0, stream>>>(q, k, v, Mtop, sc, vme, upd);
    bcast_ctx_kernel<<<(M * DM) / 256, 256, 0, stream>>>(vme, ctx);
    scatter_ctx_kernel<<<(BATCH * HEADS * UTOP * DK) / 256, 256, 0, stream>>>(upd, Mtop, ctx);

    // output projection + residual, LN1
    cvt_bf16_kernel<<<cvt_dm_blocks, 256, 0, stream>>>(ctx, cbf, (size_t)M * DM);
    gemm_wmma_bf16<<<gemm_dm_blocks, 256, 0, stream>>>(cbf, Wot[l], bo + l * DM, x, xt, nullptr, M, DM, DM, 0);
    ln_kernel<<<M, 256, 0, stream>>>(xt, ln1_g + l * DM, ln1_b + l * DM, x);

    // FFN: GELU fused into FFN1 which writes bf16 directly for FFN2
    cvt_bf16_kernel<<<cvt_dm_blocks, 256, 0, stream>>>(x, xbf, (size_t)M * DM);
    gemm_wmma_bf16<<<gemm_dff_blocks, 256, 0, stream>>>(xbf, F1t[l], ffn1_b + l * DFF, nullptr, nullptr, y1bf, M, DFF, DM, 1);
    gemm_wmma_bf16<<<gemm_dm_blocks, 256, 0, stream>>>(y1bf, F2t[l], ffn2_b + l * DM, x, xt, nullptr, M, DM, DFF, 0);
    ln_kernel<<<M, 256, 0, stream>>>(xt, ln2_g + l * DM, ln2_b + l * DM, x);
  }

  // final LN (last row only) + logits vs embedding table (first 100 rows)
  logits_kernel<<<BATCH, 256, 0, stream>>>(x, norm_g, norm_b, emb_W, out);
}
